// PPM_84490596647491
// MI455X (gfx1250) — matrix operand order repaired and checked by reading, not compiled, not run on hardware
//
#include <hip/hip_runtime.h>

// MI455X / gfx1250, wave32. bf16 WMMA with fp32 accumulation +
// async global->LDS double-buffered staging (ASYNCcnt path).
typedef __attribute__((ext_vector_type(16))) __bf16 v16bf;
typedef __attribute__((ext_vector_type(8)))  __bf16 v8bf;
typedef __attribute__((ext_vector_type(8)))  float  v8f;

#define BATCH 4
#define CCH   256
#define NPIX  4096

// ---------- Kernel 1: per-pixel channel norm, write xnT[b][n][c] bf16 ----------
__global__ void ppm_norm_kernel(const float* __restrict__ x, __bf16* __restrict__ xnT) {
    const int b = blockIdx.x >> 4;                       // 16 blocks per batch
    const int n = ((blockIdx.x & 15) << 8) + threadIdx.x;
    const float* xb = x + (size_t)b * CCH * NPIX + n;
    float ss = 0.f;
#pragma unroll 8
    for (int c = 0; c < CCH; ++c) { float v = xb[(size_t)c * NPIX]; ss += v * v; }
    const float rn = 1.0f / fmaxf(sqrtf(ss), 1e-8f);     // torch-style eps clamp
    __bf16* dst = xnT + ((size_t)b * NPIX + n) * CCH;
#pragma unroll 8
    for (int c = 0; c < CCH; ++c) dst[c] = (__bf16)(xb[(size_t)c * NPIX] * rn);
}

// ---------- Kernel 2: 1x1 conv t = W x + b in fp32, store bf16 t[b][c][n] ----------
__global__ void ppm_conv_kernel(const float* __restrict__ x, const float* __restrict__ W,
                                const float* __restrict__ bias, __bf16* __restrict__ tG) {
    int blk = blockIdx.x;
    const int nchunk = blk & 15;
    const int o = (blk >> 4) & 255;
    const int b = blk >> 12;
    const int n = (nchunk << 8) + threadIdx.x;
    const float* xb = x + (size_t)b * CCH * NPIX + n;
    const float* wr = W + (size_t)o * CCH;               // uniform per block -> scalar loads
    float acc = bias[o];
#pragma unroll 8
    for (int c = 0; c < CCH; ++c) acc = fmaf(wr[c], xb[(size_t)c * NPIX], acc);
    tG[((size_t)b * CCH + o) * NPIX + n] = (__bf16)acc;
}

// ---------- CDNA5 async global->LDS helpers ----------
__device__ __forceinline__ unsigned lds_off(const __bf16* p) {
    // generic LDS pointer: addr[31:0] is the wave-relative LDS byte address
    return (unsigned)(uintptr_t)p;
}
__device__ __forceinline__ void async_b128(const __bf16* ldsDst, const __bf16* base,
                                           unsigned byteOff) {
    asm volatile("global_load_async_to_lds_b128 %0, %1, %2"
                 :: "v"(lds_off(ldsDst)), "v"(byteOff), "s"(base)
                 : "memory");
}
__device__ __forceinline__ void wait_async0() {
    asm volatile("s_wait_asynccnt 0" ::: "memory");
}

// Issue one 64-pixel Xn tile (rows of 512B) as 8 async b128 loads per thread.
__device__ __forceinline__ void issue_xn(__bf16* dst, const __bf16* xnBatch,
                                         int pix0, int tid) {
#pragma unroll
    for (int i = 0; i < 8; ++i) {
        int idx = tid + i * 256;
        int r = idx >> 5, s = idx & 31;
        async_b128(dst + r * 264 + s * 8, xnBatch,
                   (unsigned)(((unsigned)(pix0 + r) * CCH + s * 8) * 2));
    }
}
// Issue one T^T tile: Tt[c][0..63] = t[b][c][m0..m0+63], 8 async b128 per thread.
__device__ __forceinline__ void issue_tt(__bf16* dst, const __bf16* tBatch,
                                         int m0, int tid) {
#pragma unroll
    for (int i = 0; i < 8; ++i) {
        int idx = tid + i * 256;
        int c = idx >> 3, s = idx & 7;
        async_b128(dst + c * 72 + s * 8, tBatch,
                   (unsigned)(((unsigned)c * NPIX + m0 + s * 8) * 2));
    }
}

// ---------- Kernel 3: fused  out[b,c,n] = sum_m relu(xn_n . xn_m)^2 * t[c,m] ----------
// One workgroup = (batch b, 64-pixel n-block). 8 waves = nstrip(4) x chalf(2).
// Per 64-pixel m-chunk: 32 WMMAs sim tile (K=256), relu^2 -> bf16 via LDS strip,
// 16 WMMAs accumulate OutT strip (K=64). Global->LDS staging is async,
// double-buffered so the copy of chunk mc+1 overlaps compute of chunk mc.
__launch_bounds__(256)
__global__ void ppm_fused_kernel(const __bf16* __restrict__ xnT,
                                 const __bf16* __restrict__ tG,
                                 float* __restrict__ out) {
    extern __shared__ __bf16 lds[];
    __bf16* XnBlk = lds;                       // 64 x 264
    __bf16* XnMd  = XnBlk + 64 * 264;          // 2 x (64 x 264) double buffer
    __bf16* Ttd   = XnMd  + 2 * 64 * 264;      // 2 x (256 x 72) double buffer
    __bf16* Sbuf  = Ttd   + 2 * 256 * 72;      // 8 waves x 16 x 72

    const int tid  = threadIdx.x;
    const int wave = tid >> 5;
    const int lane = tid & 31;
    const int ln   = lane & 15;
    const int h    = lane >> 4;
    const int nstrip = wave >> 1;
    const int chalf  = wave & 1;

    const int b  = blockIdx.x >> 6;
    const int nb = (blockIdx.x & 63) << 6;

    const __bf16* xnBatch = xnT + (size_t)b * NPIX * CCH;   // 2 MB per batch
    const __bf16* tBatch  = tG  + (size_t)b * CCH * NPIX;

    // Prologue: async-stage the persistent n-block tile and m-chunk 0.
    issue_xn(XnBlk, xnBatch, nb, tid);
    issue_xn(XnMd,  xnBatch, 0, tid);
    issue_tt(Ttd,   tBatch,  0, tid);

    v8f accO[8];
#pragma unroll
    for (int i = 0; i < 8; ++i) accO[i] = (v8f){0.f,0.f,0.f,0.f,0.f,0.f,0.f,0.f};

    for (int mc = 0; mc < 64; ++mc) {
        wait_async0();      // this wave's loads for chunk mc have landed in LDS
        __syncthreads();    // all waves landed + previous chunk's consumers done

        const int cur = mc & 1;
        if (mc + 1 < 64) {  // overlap next chunk's copy with this chunk's WMMAs
            const int nxt = cur ^ 1;
            issue_xn(XnMd + nxt * (64 * 264), xnBatch, (mc + 1) << 6, tid);
            issue_tt(Ttd  + nxt * (256 * 72), tBatch,  (mc + 1) << 6, tid);
        }
        const __bf16* XnM = XnMd + cur * (64 * 264);
        const __bf16* Tt  = Ttd  + cur * (256 * 72);

        // ---- sim strip: S[16 n][64 m] = relu(XnBlk_strip . XnM^T)^2 ----
        v8f sacc[4];
#pragma unroll
        for (int i = 0; i < 4; ++i) sacc[i] = (v8f){0.f,0.f,0.f,0.f,0.f,0.f,0.f,0.f};

        const __bf16* arow = XnBlk + (nstrip * 16 + ln) * 264;
#pragma unroll
        for (int kk = 0; kk < 8; ++kk) {        // K = 256 channels, 32 per step
            const int c0 = kk * 32;
            v16bf a;                            // A: elems 0-7 @ c0+h*8, 8-15 @ c0+16+h*8
            *(v8bf*)&a         = *(const v8bf*)(arow + c0 + h * 8);
            *(((v8bf*)&a) + 1) = *(const v8bf*)(arow + c0 + 16 + h * 8);
#pragma unroll
            for (int mt = 0; mt < 4; ++mt) {
                const __bf16* brow = XnM + (mt * 16 + ln) * 264 + c0 + h * 8;
                v16bf bm;                       // B: elems 0-7 @ c0+h*8, 8-15 @ c0+16+h*8
                *(v8bf*)&bm         = *(const v8bf*)(brow);
                *(((v8bf*)&bm) + 1) = *(const v8bf*)(brow + 16);
                sacc[mt] = __builtin_amdgcn_wmma_f32_16x16x32_bf16(
                    false, a, false, bm, (short)0, sacc[mt], false, false);
            }
        }

        // relu(s)^2 (gamma=2), convert to bf16 A-operand via private LDS strip
        __bf16* Sw = Sbuf + wave * (16 * 72);
#pragma unroll
        for (int mt = 0; mt < 4; ++mt) {
#pragma unroll
            for (int v = 0; v < 8; ++v) {       // D layout: row 8h+v, col ln
                float s = sacc[mt][v];
                s = fmaxf(s, 0.0f);
                s = s * s;
                Sw[(8 * h + v) * 72 + mt * 16 + ln] = (__bf16)s;
            }
        }
        // same-wave DS ops complete in order -> strip readable below without barrier

        // ---- OutT strip accumulate: accO += S(16x64) @ Tt(64x128-half) ----
#pragma unroll
        for (int ks = 0; ks < 2; ++ks) {        // K = 64 m, 32 per step
            const __bf16* ar = Sw + ln * 72 + ks * 32;
            v16bf a2;
            *(v8bf*)&a2         = *(const v8bf*)(ar + h * 8);
            *(((v8bf*)&a2) + 1) = *(const v8bf*)(ar + 16 + h * 8);
#pragma unroll
            for (int ct = 0; ct < 8; ++ct) {
                const __bf16* br = Tt + (chalf * 128 + ct * 16 + ln) * 72 + ks * 32 + h * 8;
                v16bf b2;
                *(v8bf*)&b2         = *(const v8bf*)(br);
                *(((v8bf*)&b2) + 1) = *(const v8bf*)(br + 16);
                accO[ct] = __builtin_amdgcn_wmma_f32_16x16x32_bf16(
                    false, a2, false, b2, (short)0, accO[ct], false, false);
            }
        }
    }

    // ---- epilogue: out[b][c][n], lane's 8 accum values are 8 consecutive n ----
#pragma unroll
    for (int ct = 0; ct < 8; ++ct) {
        const int c  = chalf * 128 + ct * 16 + ln;
        const int n0 = nb + nstrip * 16 + 8 * h;
        float* op = out + ((size_t)b * CCH + c) * NPIX + n0;
        float4* pv = (float4*)&accO[ct];
        *(float4*)(op)     = pv[0];
        *(float4*)(op + 4) = pv[1];
    }
}

extern "C" void kernel_launch(void* const* d_in, const int* in_sizes, int n_in,
                              void* d_out, int out_size, void* d_ws, size_t ws_size,
                              hipStream_t stream) {
    const float* x    = (const float*)d_in[0];   // [4,256,64,64]
    const float* W    = (const float*)d_in[1];   // [256,256]
    const float* bias = (const float*)d_in[2];   // [256]
    float* out = (float*)d_out;

    __bf16* xnT = reinterpret_cast<__bf16*>(d_ws);                       // 8.39 MB
    __bf16* tG  = reinterpret_cast<__bf16*>((char*)d_ws +
                      (size_t)BATCH * NPIX * CCH * sizeof(__bf16));      // 8.39 MB

    ppm_norm_kernel<<<BATCH * (NPIX / 256), 256, 0, stream>>>(x, xnT);
    ppm_conv_kernel<<<BATCH * CCH * (NPIX / 256), 256, 0, stream>>>(x, W, bias, tG);

    const size_t ldsBytes =
        (size_t)(64 * 264 + 2 * 64 * 264 + 2 * 256 * 72 + 8 * 16 * 72) * sizeof(__bf16); // ~189 KB
    ppm_fused_kernel<<<BATCH * (NPIX / 64), 256, ldsBytes, stream>>>(xnT, tG, out);
}
